// Encoder_88235808129076
// MI455X (gfx1250) — matrix operand order repaired and checked by reading, not compiled, not run on hardware
//
#include <hip/hip_runtime.h>
#include <hip/hip_bf16.h>

// ---- problem constants (match reference) ----
#define BB 128
#define TT 64
#define II 128
#define EE 256
#define HH 512

typedef __attribute__((ext_vector_type(16))) _Float16 v16h;
typedef __attribute__((ext_vector_type(8)))  float    v8f;

// ------------------------------------------------------------------
// WMMA helpers: D = A(16x32 f16) * B(32x16 f16) + C(16x16 f32), wave32
// ------------------------------------------------------------------
__device__ __forceinline__ v8f wmma_f16(v16h a, v16h b, v8f c) {
  return __builtin_amdgcn_wmma_f32_16x16x32_f16(
      /*neg_a=*/false, a, /*neg_b=*/false, b,
      /*c_mod=*/(short)0, c, /*reuse_a=*/false, /*reuse_b=*/false);
}

// A fragment from row-major [M][K] f16 matrix.
// lane<16: M=m0+lane,   K = k0 + {0..7, 16..23}
// lane>=16:M=m0+lane-16,K = k0 + {8..15,24..31}
__device__ __forceinline__ v16h load_A(const _Float16* __restrict__ A, int lda,
                                       int m0, int k0, int lane) {
  int half = lane >> 4, row = lane & 15;
  const _Float16* p = A + (size_t)(m0 + row) * lda + k0 + half * 8;
  union { v16h v; float4 f[2]; } u;
  u.f[0] = *reinterpret_cast<const float4*>(p);        // 8 f16 = 16 B
  u.f[1] = *reinterpret_cast<const float4*>(p + 16);   // next K-group
  return u.v;
}

// B fragment from row-major [N][K] f16 matrix (i.e. W[j][k] directly).
// lane&15 = N-col; lane<16: K = k0 + {0..7,16..23}; lane>=16: K = k0 + {8..15,24..31}.
__device__ __forceinline__ v16h load_B(const _Float16* __restrict__ W, int ldb,
                                       int n0, int k0, int lane) {
  int half = lane >> 4, col = lane & 15;
  const _Float16* p = W + (size_t)(n0 + col) * ldb + k0 + half * 8;
  union { v16h v; float4 f[2]; } u;
  u.f[0] = *reinterpret_cast<const float4*>(p);
  u.f[1] = *reinterpret_cast<const float4*>(p + 16);
  return u.v;
}

__device__ __forceinline__ float sigmoidf_(float x) {
  return 1.0f / (1.0f + expf(-x));
}

// ------------------------------------------------------------------
// Kernel 0: weight conversion f32->f16 + state zero-init
// ------------------------------------------------------------------
__global__ void prep_kernel(const float* __restrict__ Wih, const float* __restrict__ Whh,
                            const float* __restrict__ a2w,
                            _Float16* __restrict__ Wih16, _Float16* __restrict__ Whh16,
                            _Float16* __restrict__ a2w16,
                            float* __restrict__ hbuf, float* __restrict__ cbuf,
                            _Float16* __restrict__ h16a, _Float16* __restrict__ h16b,
                            _Float16* __restrict__ w16) {
  int n = blockIdx.x * blockDim.x + threadIdx.x;
  if (n < 4 * HH * HH) Whh16[n] = (_Float16)Whh[n];      // 1,048,576
  if (n < 4 * HH * II) Wih16[n] = (_Float16)Wih[n];      // 262,144
  if (n < TT * TT)     a2w16[n] = (_Float16)a2w[n];      // 4,096
  if (n < BB * HH) {
    hbuf[n] = 0.0f; cbuf[n] = 0.0f;
    h16a[n] = (_Float16)0.0f; h16b[n] = (_Float16)0.0f;
  }
  if (n < BB * II) w16[n] = (_Float16)0.0f;
}

// ------------------------------------------------------------------
// Kernel 1: news_linear. One wave per output (b,t,i); E=256 via 2x float4/lane.
// Streams the 1.07 GB input once -> ~46 us HBM floor for the whole op.
// Writes x f32 [B][T][I] and xt16 f16 [B][I][T] (transposed, for z2 WMMA A-op).
// ------------------------------------------------------------------
__global__ void news_kernel(const float* __restrict__ in, const float* __restrict__ nW,
                            const float* __restrict__ nb,
                            float* __restrict__ x, _Float16* __restrict__ xt16) {
  int wid  = (blockIdx.x * blockDim.x + threadIdx.x) >> 5;
  int lane = threadIdx.x & 31;
  if (wid >= BB * TT * II) return;
  const float4* p = reinterpret_cast<const float4*>(in + (size_t)wid * EE) + lane * 2;
  const float4* q = reinterpret_cast<const float4*>(nW) + lane * 2;
  float4 a0 = p[0], a1 = p[1];
  float4 w0 = q[0], w1 = q[1];
  float s = a0.x * w0.x + a0.y * w0.y + a0.z * w0.z + a0.w * w0.w +
            a1.x * w1.x + a1.y * w1.y + a1.z * w1.z + a1.w * w1.w;
  for (int m = 16; m; m >>= 1) s += __shfl_xor(s, m, 32);
  if (lane == 0) {
    s += nb[0];
    x[wid] = s;
    int i  = wid % II;
    int bt = wid / II;
    int tt = bt % TT;
    int b  = bt / TT;
    xt16[((size_t)b * II + i) * TT + tt] = (_Float16)s;
  }
}

// ------------------------------------------------------------------
// Kernel 2: z2[b,i,s] = sum_t x[b,t,i] * A2[s,t] + b2[s]   (loop invariant)
// One block per b (8 waves); wave = M-tile; 4 N-tiles x 2 K-chunks unrolled.
// ------------------------------------------------------------------
__global__ void z2_kernel(const _Float16* __restrict__ xt16,
                          const _Float16* __restrict__ a2w16,
                          const float* __restrict__ a2b,
                          float* __restrict__ z2) {
  int b    = blockIdx.x;
  int wave = threadIdx.x >> 5;
  int lane = threadIdx.x & 31;
  const _Float16* A = xt16 + (size_t)b * II * TT;   // [128][64] row-major
  int m0 = wave * 16;
#pragma unroll
  for (int nt = 0; nt < 4; ++nt) {
    int n0 = nt * 16;
    v8f acc = {};
#pragma unroll
    for (int kc = 0; kc < TT; kc += 32) {
      v16h af = load_A(A, TT, m0, kc, lane);
      v16h bf = load_B(a2w16, TT, n0, kc, lane);   // A2 is [s][t] = [N][K]
      acc = wmma_f16(af, bf, acc);
    }
    int half = lane >> 4, col = lane & 15;
    int n = n0 + col;
    float bias = a2b[n];
#pragma unroll
    for (int r = 0; r < 8; ++r) {
      int m = m0 + r + half * 8;
      z2[((size_t)(b * II + m)) * TT + n] = acc[r] + bias;
    }
  }
}

// ------------------------------------------------------------------
// Kernel 3 (per step): z1 GEMV + tanh-attention + softmax + w = attn*x_t
// One block per b, 128 threads (4 waves). h/c rows hoisted to registers.
// ------------------------------------------------------------------
__global__ void attn_kernel(int t, const float* __restrict__ x, const float* __restrict__ z2,
                            const float* __restrict__ hbuf, const float* __restrict__ cbuf,
                            const float* __restrict__ a1w, const float* __restrict__ a1b,
                            const float* __restrict__ a3w, const float* __restrict__ a3b,
                            _Float16* __restrict__ w16) {
  int b    = blockIdx.x;
  int tid  = threadIdx.x;
  int wave = tid >> 5, lane = tid & 31;
  __shared__ float z1s[TT];
  __shared__ float red[8];

  // hoist h[b,:], c[b,:] into registers: lane covers k = lane + 32*j
  const float* hb = hbuf + (size_t)b * HH;
  const float* cb = cbuf + (size_t)b * HH;
  float hr[16], cr[16];
#pragma unroll
  for (int j = 0; j < 16; ++j) {
    hr[j] = hb[lane + 32 * j];
    cr[j] = cb[lane + 32 * j];
  }
  // z1[t'] = [h,c] . attn1_W[t',:]   (K = 2H = 1024)
  for (int tt = wave; tt < TT; tt += 4) {
    const float* a1 = a1w + (size_t)tt * (2 * HH);
    float s = 0.0f;
#pragma unroll
    for (int j = 0; j < 16; ++j) {
      int k = lane + 32 * j;
      s += hr[j] * a1[k] + cr[j] * a1[HH + k];
    }
    for (int m = 16; m; m >>= 1) s += __shfl_xor(s, m, 32);
    if (lane == 0) z1s[tt] = s + a1b[tt];
  }
  __syncthreads();

  // thread == driver index i
  int i = tid;
  const float* z2r = z2 + ((size_t)(b * II + i)) * TT;
  float z3 = a3b[0];
#pragma unroll 4
  for (int tt = 0; tt < TT; ++tt)
    z3 += tanhf(z1s[tt] + z2r[tt]) * a3w[tt];

  // softmax over i (block-wide, 128 lanes)
  float mx = z3;
  for (int m = 16; m; m >>= 1) mx = fmaxf(mx, __shfl_xor(mx, m, 32));
  if (lane == 0) red[wave] = mx;
  __syncthreads();
  mx = fmaxf(fmaxf(red[0], red[1]), fmaxf(red[2], red[3]));
  float e = expf(z3 - mx);
  float sm = e;
  for (int m = 16; m; m >>= 1) sm += __shfl_xor(sm, m, 32);
  __syncthreads();
  if (lane == 0) red[4 + wave] = sm;
  __syncthreads();
  sm = red[4] + red[5] + red[6] + red[7];

  float wv = (e / sm) * x[((size_t)b * TT + t) * II + i];
  w16[(size_t)b * II + i] = (_Float16)wv;
}

// ------------------------------------------------------------------
// Kernel 4 (per step): gates = w@W_ih^T + h@W_hh^T + b, fused LSTM update.
// K-split across wave pairs: 512 waves (2x occupancy), 10 chunks / 40 WMMAs
// per wave, FULLY UNROLLED (compile-time trip counts; p*32 folds into the
// base address so all loads are base+immediate b128 in s_clause groups).
// Partials combined through 16 KB of LDS.
// ------------------------------------------------------------------
__global__ void gates_kernel(int t, const _Float16* __restrict__ w16,
                             const _Float16* __restrict__ h16in,
                             const _Float16* __restrict__ Wih, const _Float16* __restrict__ Whh,
                             const float* __restrict__ bih, const float* __restrict__ bhh,
                             float* __restrict__ hbuf, float* __restrict__ cbuf,
                             _Float16* __restrict__ h16out, float* __restrict__ out) {
  __shared__ float part[4][4][8][32];   // [pair-in-block][gate][row][lane]

  int wave = threadIdx.x >> 5;                       // 0..7
  int lane = threadIdx.x & 31;
  int gw   = blockIdx.x * 8 + wave;                  // 0..511
  int pair = gw >> 1;                                // 0..255 tile-group
  int p    = gw & 1;                                 // K-split half
  int pl   = wave >> 1;                              // pair index within block
  int mt = pair >> 5;          // 0..7   (M tile: batch rows)
  int nt = pair & 31;          // 0..31  (N tile within H)
  int m0 = mt * 16, n0 = nt * 16;
  int kbase = p * 32;          // uniform: folds into base addresses

  v8f acc0 = {}, acc1 = {}, acc2 = {}, acc3 = {};

  // input contribution: A = w16 [128 x 128], B = W_ih [2048 x 128] ([N][K])
#pragma unroll
  for (int kk = 0; kk < 2; ++kk) {
    int kc = kbase + kk * 64;
    v16h a  = load_A(w16, II, m0, kc, lane);
    v16h b0 = load_B(Wih, II, n0 + 0 * HH, kc, lane);
    v16h b1 = load_B(Wih, II, n0 + 1 * HH, kc, lane);
    v16h b2 = load_B(Wih, II, n0 + 2 * HH, kc, lane);
    v16h b3 = load_B(Wih, II, n0 + 3 * HH, kc, lane);
    acc0 = wmma_f16(a, b0, acc0);
    acc1 = wmma_f16(a, b1, acc1);
    acc2 = wmma_f16(a, b2, acc2);
    acc3 = wmma_f16(a, b3, acc3);
  }
  // hidden contribution: A = h16 [128 x 512], B = W_hh [2048 x 512] ([N][K])
#pragma unroll
  for (int kk = 0; kk < 8; ++kk) {
    int kc = kbase + kk * 64;
    v16h a  = load_A(h16in, HH, m0, kc, lane);
    v16h b0 = load_B(Whh, HH, n0 + 0 * HH, kc, lane);
    v16h b1 = load_B(Whh, HH, n0 + 1 * HH, kc, lane);
    v16h b2 = load_B(Whh, HH, n0 + 2 * HH, kc, lane);
    v16h b3 = load_B(Whh, HH, n0 + 3 * HH, kc, lane);
    acc0 = wmma_f16(a, b0, acc0);
    acc1 = wmma_f16(a, b1, acc1);
    acc2 = wmma_f16(a, b2, acc2);
    acc3 = wmma_f16(a, b3, acc3);
  }

  // combine K-split partials through LDS (odd wave -> even wave)
  if (p) {
#pragma unroll
    for (int r = 0; r < 8; ++r) {
      part[pl][0][r][lane] = acc0[r];
      part[pl][1][r][lane] = acc1[r];
      part[pl][2][r][lane] = acc2[r];
      part[pl][3][r][lane] = acc3[r];
    }
  }
  __syncthreads();
  if (p) return;

  // fused bias + LSTM pointwise update (even wave of each pair)
  int half = lane >> 4, col = lane & 15;
  int n = n0 + col;
  float bi = bih[n + 0 * HH] + bhh[n + 0 * HH];
  float bf = bih[n + 1 * HH] + bhh[n + 1 * HH];
  float bg = bih[n + 2 * HH] + bhh[n + 2 * HH];
  float bo = bih[n + 3 * HH] + bhh[n + 3 * HH];
#pragma unroll
  for (int r = 0; r < 8; ++r) {
    int m = m0 + r + half * 8;                 // batch row
    float ig = sigmoidf_(acc0[r] + part[pl][0][r][lane] + bi);
    float fg = sigmoidf_(acc1[r] + part[pl][1][r][lane] + bf);
    float gg = tanhf   (acc2[r] + part[pl][2][r][lane] + bg);
    float og = sigmoidf_(acc3[r] + part[pl][3][r][lane] + bo);
    size_t hc_idx = (size_t)m * HH + n;
    float cnew = fg * cbuf[hc_idx] + ig * gg;
    float hnew = og * tanhf(cnew);
    cbuf[hc_idx]   = cnew;
    hbuf[hc_idx]   = hnew;
    h16out[hc_idx] = (_Float16)hnew;
    out[((size_t)m * TT + t) * HH + n] = hnew;   // hs[b][t][h]
  }
}

// ------------------------------------------------------------------
// Host-side orchestration
// ------------------------------------------------------------------
extern "C" void kernel_launch(void* const* d_in, const int* in_sizes, int n_in,
                              void* d_out, int out_size, void* d_ws, size_t ws_size,
                              hipStream_t stream) {
  (void)in_sizes; (void)n_in; (void)out_size; (void)ws_size;

  const float* input_data = (const float*)d_in[0];   // [B,T,I,E]
  const float* news_W     = (const float*)d_in[1];   // [1,E]
  const float* news_b     = (const float*)d_in[2];   // [1]
  const float* attn1_W    = (const float*)d_in[3];   // [T,2H]
  const float* attn1_b    = (const float*)d_in[4];   // [T]
  const float* attn2_W    = (const float*)d_in[5];   // [T,T]
  const float* attn2_b    = (const float*)d_in[6];   // [T]
  const float* attn3_W    = (const float*)d_in[7];   // [1,T]
  const float* attn3_b    = (const float*)d_in[8];   // [1]
  const float* W_ih       = (const float*)d_in[9];   // [4H,I]
  const float* W_hh       = (const float*)d_in[10];  // [4H,H]
  const float* b_ih       = (const float*)d_in[11];  // [4H]
  const float* b_hh       = (const float*)d_in[12];  // [4H]
  float* out = (float*)d_out;                        // [B,T,H]

  // ---- workspace carve-up (256B aligned) ----
  char* ws = (char*)d_ws;
  size_t off = 0;
  auto carve = [&](size_t bytes) { char* p = ws + off; off = (off + bytes + 255) & ~(size_t)255; return p; };
  float*    x      = (float*)   carve((size_t)BB * TT * II * 4);   // 4 MB
  _Float16* xt16   = (_Float16*)carve((size_t)BB * II * TT * 2);   // 2 MB
  float*    z2     = (float*)   carve((size_t)BB * II * TT * 4);   // 4 MB
  _Float16* Wih16  = (_Float16*)carve((size_t)4 * HH * II * 2);    // 512 KB
  _Float16* Whh16  = (_Float16*)carve((size_t)4 * HH * HH * 2);    // 2 MB
  _Float16* a2w16  = (_Float16*)carve((size_t)TT * TT * 2);        // 8 KB
  float*    hbuf   = (float*)   carve((size_t)BB * HH * 4);        // 256 KB
  float*    cbuf   = (float*)   carve((size_t)BB * HH * 4);        // 256 KB
  _Float16* h16a   = (_Float16*)carve((size_t)BB * HH * 2);        // 128 KB
  _Float16* h16b   = (_Float16*)carve((size_t)BB * HH * 2);        // 128 KB
  _Float16* w16    = (_Float16*)carve((size_t)BB * II * 2);        // 32 KB

  // 0) convert weights, zero state
  prep_kernel<<<4096, 256, 0, stream>>>(W_ih, W_hh, attn2_W, Wih16, Whh16, a2w16,
                                        hbuf, cbuf, h16a, h16b, w16);
  // 1) news_linear (streams the 1.07 GB input once)
  {
    int nwaves = BB * TT * II;               // one wave per output
    int blocks = (nwaves * 32) / 256;        // 131072 blocks
    news_kernel<<<blocks, 256, 0, stream>>>(input_data, news_W, news_b, x, xt16);
  }
  // 2) loop-invariant z2 via WMMA
  z2_kernel<<<BB, 256, 0, stream>>>(xt16, a2w16, attn2_b, z2);

  // 3) sequential LSTM scan: 64 steps x (attention, gates+update)
  for (int t = 0; t < TT; ++t) {
    attn_kernel<<<BB, 128, 0, stream>>>(t, x, z2, hbuf, cbuf,
                                        attn1_W, attn1_b, attn3_W, attn3_b, w16);
    const _Float16* hin = (t & 1) ? h16b : h16a;
    _Float16*       hout = (t & 1) ? h16a : h16b;
    gates_kernel<<<64, 256, 0, stream>>>(t, w16, hin, Wih16, Whh16,
                                         b_ih, b_hh, hbuf, cbuf, hout, out);
  }
}
